// GCN_27960237097168
// MI455X (gfx1250) — matrix operand order repaired and checked by reading, not compiled, not run on hardware
//
#include <hip/hip_runtime.h>
#include <math.h>

typedef __attribute__((ext_vector_type(16))) _Float16 v16h;
typedef __attribute__((ext_vector_type(8)))  _Float16 v8h;
typedef __attribute__((ext_vector_type(8)))  float    v8f;

#define Hdim 128
#define LDS_PITCH 136   // 128 + 8 halves pad: 272B row stride -> 4-bank rotation/lane

// ---------------------------------------------------------------------------
// Transpose + convert the three 128x128 f32 weight matrices into f16 WT[n][k]
// so WMMA B-fragments are contiguous in k.
// ---------------------------------------------------------------------------
__global__ __launch_bounds__(256) void gcn_cvt_wt(const float* __restrict__ W1,
                                                  const float* __restrict__ W2,
                                                  const float* __restrict__ W3,
                                                  _Float16* __restrict__ WT) {
  int i = blockIdx.x * 256 + threadIdx.x;       // 0 .. 16383
  int layer = blockIdx.y;                       // 0..2
  const float* W = (layer == 0) ? W1 : (layer == 1) ? W2 : W3;
  int n = i >> 7;          // output channel
  int k = i & 127;         // input channel
  WT[(size_t)layer * Hdim * Hdim + (size_t)n * Hdim + k] = (_Float16)W[k * Hdim + n];
}

// ---------------------------------------------------------------------------
// Degree: deg[n] = 1 (self loop) + #edges with dst == n; then dinv = rsqrt.
// ---------------------------------------------------------------------------
__global__ void gcn_deg_init(float* __restrict__ deg, int N) {
  int i = blockIdx.x * 256 + threadIdx.x;
  if (i < N) deg[i] = 1.0f;
}
__global__ void gcn_deg_accum(const int* __restrict__ dst, float* __restrict__ deg, int E) {
  int e = blockIdx.x * 256 + threadIdx.x;
  if (e < E) atomicAdd(&deg[dst[e]], 1.0f);
}
__global__ void gcn_deg_rsqrt(float* __restrict__ deg, int N) {
  int i = blockIdx.x * 256 + threadIdx.x;
  if (i < N) {
    float d = deg[i];
    deg[i] = rsqrtf(d > 1.0f ? d : 1.0f);     // in-place: becomes dinv
  }
}

// ---------------------------------------------------------------------------
// WMMA GEMM: H[N x 128] = X[N x 128] (f32, row major) * W (WT f16 [n][k]).
// Block = 256 threads = 8 waves. k 0..63 of each WT row (16KB f16) is staged in padded LDS once
// per block. Each wave owns a DISTINCT 16-row stripe and computes all 128
// output columns: 8 v8f accumulators, A fragment loaded once per K-step and
// reused by 8 WMMAs -> X is read exactly once across the whole GEMM.
// Wave-uniform row guard after the barrier keeps EXEC all-ones for WMMA.
// ---------------------------------------------------------------------------
__global__ __launch_bounds__(256) void gcn_gemm_wmma(const float* __restrict__ X,
                                                     const _Float16* __restrict__ WT,
                                                     float* __restrict__ Hout, int N) {
  __shared__ _Float16 wlds[Hdim * LDS_PITCH];   // 34.8 KB

  const int tid  = threadIdx.x;
  const int lane = tid & 31;
  const int wave = tid >> 5;                    // 0..7 -> row stripe
  const int hi   = lane >> 4;                   // 0 | 1 (lane half)
  const int l15  = lane & 15;

  // ---- stage WT into LDS: 1024 chunks of 16B, 256 threads ----
#pragma unroll
  for (int c = tid; c < Hdim * 8; c += 256) {   // chunk = (row, seg of 8 halves)
    int row = c >> 3, seg = c & 7;
    const uint4* g = (const uint4*)(WT + (size_t)row * Hdim + seg * 8);
    *(uint4*)(&wlds[row * LDS_PITCH + seg * 8]) = *g;
  }
  __syncthreads();

  const int row0 = blockIdx.x * 128 + wave * 16;
  if (row0 < N) {                               // wave-uniform guard
    const int r = row0 + l15;                   // A row for this lane
    const int kbase_a = hi * 8;                 // A: lanes 0-15 K+0, 16-31 K+8
    const int kbase_b = hi * 8;                 // B: lanes 0-15 K+0, 16-31 K+8 (and +16)
    const float* xrow = X + (size_t)r * Hdim;

    v8f acc[8] = {};

#pragma unroll
    for (int kk = 0; kk < Hdim; kk += 32) {
      // ---- A fragment: f32 -> f16, two contiguous 16B runs ----
      const float4* xp0 = (const float4*)(xrow + kk + kbase_a);        // K..K+7
      const float4* xp1 = (const float4*)(xrow + kk + kbase_a + 16);   // K+16..K+23
      float4 x0 = xp0[0], x1 = xp0[1];
      float4 x2 = xp1[0], x3 = xp1[1];
      v16h a;
      a[0]  = (_Float16)x0.x; a[1]  = (_Float16)x0.y;
      a[2]  = (_Float16)x0.z; a[3]  = (_Float16)x0.w;
      a[4]  = (_Float16)x1.x; a[5]  = (_Float16)x1.y;
      a[6]  = (_Float16)x1.z; a[7]  = (_Float16)x1.w;
      a[8]  = (_Float16)x2.x; a[9]  = (_Float16)x2.y;
      a[10] = (_Float16)x2.z; a[11] = (_Float16)x2.w;
      a[12] = (_Float16)x3.x; a[13] = (_Float16)x3.y;
      a[14] = (_Float16)x3.z; a[15] = (_Float16)x3.w;

      // ---- sweep 8 column tiles; B fragments from LDS ----
#pragma unroll
      for (int ct = 0; ct < 8; ++ct) {
        const int n = ct * 16 + l15;            // B column for this lane
        const v8h* wp = (const v8h*)(&wlds[n * LDS_PITCH + kk + kbase_b]);
        v8h b0 = wp[0], b1 = wp[2];
        v16h b = __builtin_shufflevector(b0, b1, 0, 1, 2, 3, 4, 5, 6, 7,
                                                 8, 9, 10, 11, 12, 13, 14, 15);
        acc[ct] = __builtin_amdgcn_wmma_f32_16x16x32_f16(
            false, a, false, b, (short)0, acc[ct], false, false);
      }
    }

    // ---- store: VGPR j -> row (hi*8 + j), col (ct*16 + l15) ----
#pragma unroll
    for (int ct = 0; ct < 8; ++ct) {
      float* out = Hout + (size_t)(row0 + hi * 8) * Hdim + ct * 16 + l15;
#pragma unroll
      for (int j = 0; j < 8; ++j) out[(size_t)j * Hdim] = acc[ct][j];
    }
  }
}

// ---------------------------------------------------------------------------
// Zero the aggregation buffer.
// ---------------------------------------------------------------------------
__global__ void gcn_zero(float* __restrict__ p, int count) {
  int i = blockIdx.x * 256 + threadIdx.x;
  if (i < count) p[i] = 0.0f;
}

// ---------------------------------------------------------------------------
// Edge aggregation: agg[dst] += h[src] * dinv[src]*dinv[dst].
// One 32-lane group per edge; each lane handles a float4 (128 feats = 32x4).
// ---------------------------------------------------------------------------
__global__ __launch_bounds__(256) void gcn_aggregate(const float* __restrict__ h,
                                                     const int* __restrict__ src,
                                                     const int* __restrict__ dst,
                                                     const float* __restrict__ dinv,
                                                     float* __restrict__ agg, int E) {
  int e = blockIdx.x * 8 + (threadIdx.x >> 5);
  if (e >= E) return;
  int lane = threadIdx.x & 31;
  int s = src[e], d = dst[e];
  float nrm = dinv[s] * dinv[d];
  float4 v = ((const float4*)(h + (size_t)s * Hdim))[lane];
  float* o = agg + (size_t)d * Hdim + lane * 4;
  atomicAdd(o + 0, v.x * nrm);
  atomicAdd(o + 1, v.y * nrm);
  atomicAdd(o + 2, v.z * nrm);
  atomicAdd(o + 3, v.w * nrm);
}

// ---------------------------------------------------------------------------
// Epilogue: fold in self-loop (h*dinv^2), bias, ReLU. In-place on agg.
// ---------------------------------------------------------------------------
__global__ void gcn_finish(float* __restrict__ agg, const float* __restrict__ h,
                           const float* __restrict__ dinv,
                           const float* __restrict__ bias, int total) {
  int i = blockIdx.x * 256 + threadIdx.x;
  if (i >= total) return;
  int node = i >> 7;
  int f    = i & 127;
  float di = dinv[node];
  float v  = agg[i] + h[i] * di * di + bias[f];
  agg[i] = v > 0.0f ? v : 0.0f;
}

// ---------------------------------------------------------------------------
// Readout: sigmoid(x[n] . Wout + bout), one 128-thread block per node.
// ---------------------------------------------------------------------------
__global__ __launch_bounds__(128) void gcn_readout(const float* __restrict__ x,
                                                   const float* __restrict__ Wout,
                                                   const float* __restrict__ bout,
                                                   float* __restrict__ out, int N) {
  __shared__ float red[128];
  int n = blockIdx.x;
  int t = threadIdx.x;
  red[t] = x[(size_t)n * Hdim + t] * Wout[t];
  __syncthreads();
#pragma unroll
  for (int s = 64; s > 0; s >>= 1) {
    if (t < s) red[t] += red[t + s];
    __syncthreads();
  }
  if (t == 0) out[n] = 1.0f / (1.0f + expf(-(red[0] + bout[0])));
}

// ---------------------------------------------------------------------------
// Launch
// ---------------------------------------------------------------------------
extern "C" void kernel_launch(void* const* d_in, const int* in_sizes, int n_in,
                              void* d_out, int out_size, void* d_ws, size_t ws_size,
                              hipStream_t stream) {
  const float* x     = (const float*)d_in[0];
  const int*   eidx  = (const int*)d_in[1];
  const float* W1    = (const float*)d_in[2];
  const float* b1    = (const float*)d_in[3];
  const float* W2    = (const float*)d_in[4];
  const float* b2    = (const float*)d_in[5];
  const float* W3    = (const float*)d_in[6];
  const float* b3    = (const float*)d_in[7];
  const float* Wout  = (const float*)d_in[8];
  const float* bout  = (const float*)d_in[9];
  float*       out   = (float*)d_out;

  const int N = in_sizes[0] / Hdim;   // 50000
  const int E = in_sizes[1] / 2;      // 600000
  const int* src = eidx;
  const int* dst = eidx + E;

  // Workspace layout (base assumed >= 256B aligned by harness)
  char* ws = (char*)d_ws;
  _Float16* WT   = (_Float16*)ws;                       // 3*128*128*2 = 98304 B
  size_t off = 98304;
  float* dinv = (float*)(ws + off);  off += (size_t)N * sizeof(float);       // 200000 B
  off = (off + 63) & ~(size_t)63;
  float* hbuf = (float*)(ws + off);  off += (size_t)N * Hdim * sizeof(float);
  off = (off + 63) & ~(size_t)63;
  float* agg  = (float*)(ws + off);                      // doubles as layer input
  (void)ws_size; (void)n_in; (void)out_size;

  const int total = N * Hdim;

  // Weight transpose+convert (every call; deterministic)
  dim3 gw(64, 3);
  gcn_cvt_wt<<<gw, 256, 0, stream>>>(W1, W2, W3, WT);

  // Degrees -> dinv
  gcn_deg_init<<<(N + 255) / 256, 256, 0, stream>>>(dinv, N);
  gcn_deg_accum<<<(E + 255) / 256, 256, 0, stream>>>(dst, dinv, E);
  gcn_deg_rsqrt<<<(N + 255) / 256, 256, 0, stream>>>(dinv, N);

  const float* biases[3] = { b1, b2, b3 };
  for (int layer = 0; layer < 3; ++layer) {
    const float* xin = (layer == 0) ? x : agg;
    const _Float16* WTl = WT + (size_t)layer * Hdim * Hdim;

    gcn_gemm_wmma<<<(N + 127) / 128, 256, 0, stream>>>(xin, WTl, hbuf, N);
    gcn_zero<<<(total + 255) / 256, 256, 0, stream>>>(agg, total);
    gcn_aggregate<<<(E + 7) / 8, 256, 0, stream>>>(hbuf, src, dst, dinv, agg, E);
    gcn_finish<<<(total + 255) / 256, 256, 0, stream>>>(agg, hbuf, dinv, biases[layer], total);
  }

  gcn_readout<<<N, 128, 0, stream>>>(agg, Wout, bout, out, N);
}
